// MultiHeadRelationalModuleImage_43791486550346
// MI455X (gfx1250) — matrix operand order repaired and checked by reading, not compiled, not run on hardware
//
#include <hip/hip_runtime.h>

typedef __attribute__((ext_vector_type(16))) __bf16       v16bf;
typedef __attribute__((ext_vector_type(8)))  float        v8f;
typedef __attribute__((ext_vector_type(4)))  unsigned int u32x4;

#define BB   8
#define NNODE 3600
#define NPAD 3616      // 3600 padded to multiple of 32 for WMMA K-loops
#define DDIM 64
#define KCAT 128
#define EPSV 1e-5f

// ---------- scalar bf16 helpers (round-to-nearest-even) ----------
__device__ __forceinline__ unsigned short f2bf(float x) {
  unsigned u = __float_as_uint(x);
  unsigned r = 0x7FFFu + ((u >> 16) & 1u);
  return (unsigned short)((u + r) >> 16);
}
__device__ __forceinline__ float bf2f(unsigned short h) {
  return __uint_as_float(((unsigned)h) << 16);
}
__device__ __forceinline__ float eluf(float x) { return x > 0.f ? x : __expf(x) - 1.f; }

// =============== conv1: [8,3,64,64] -> relu -> [8,8,60,60] ===============
__global__ void conv1_kernel(const float* __restrict__ x, const float* __restrict__ w,
                             const float* __restrict__ b, float* __restrict__ out) {
  __shared__ float sw[1184];                      // 8*3*49 weights + 8 bias
  int tid = threadIdx.x;
  for (int i = tid; i < 1184; i += 256) sw[i] = (i < 1176) ? w[i] : b[i - 1176];
  __syncthreads();
  int idx = blockIdx.x * 256 + tid;
  if (idx >= BB * 3600) return;
  int bb = idx / 3600, p = idx % 3600, oy = p / 60, ox = p % 60;
  float acc[8];
#pragma unroll
  for (int oc = 0; oc < 8; oc++) acc[oc] = sw[1176 + oc];
  for (int c = 0; c < 3; c++)
    for (int ky = 0; ky < 7; ky++) {
      int iy = oy + ky - 1;
      if (iy < 0 || iy >= 64) continue;
      for (int kx = 0; kx < 7; kx++) {
        int ix = ox + kx - 1;
        if (ix < 0 || ix >= 64) continue;
        float xv = x[((bb * 3 + c) * 64 + iy) * 64 + ix];
#pragma unroll
        for (int oc = 0; oc < 8; oc++)
          acc[oc] = fmaf(xv, sw[((oc * 3 + c) * 7 + ky) * 7 + kx], acc[oc]);
      }
    }
#pragma unroll
  for (int oc = 0; oc < 8; oc++)
    out[((bb * 8 + oc) * 60 + oy) * 60 + ox] = fmaxf(acc[oc], 0.f);
}

// ====== conv2 + relu + coord channels -> feats [8,3600,12] f32 ======
__global__ void conv2_feats_kernel(const float* __restrict__ h1, const float* __restrict__ w,
                                   const float* __restrict__ b, float* __restrict__ feats) {
  __shared__ float sw[730];                       // 10*8*9 weights + 10 bias
  int tid = threadIdx.x;
  for (int i = tid; i < 730; i += 256) sw[i] = (i < 720) ? w[i] : b[i - 720];
  __syncthreads();
  int idx = blockIdx.x * 256 + tid;
  if (idx >= BB * 3600) return;
  int bb = idx / 3600, p = idx % 3600, oy = p / 60, ox = p % 60;
  float acc[10];
#pragma unroll
  for (int oc = 0; oc < 10; oc++) acc[oc] = sw[720 + oc];
  for (int c = 0; c < 8; c++)
    for (int ky = 0; ky < 3; ky++) {
      int iy = oy + ky - 1;
      if (iy < 0 || iy >= 60) continue;
      for (int kx = 0; kx < 3; kx++) {
        int ix = ox + kx - 1;
        if (ix < 0 || ix >= 60) continue;
        float v = h1[((bb * 8 + c) * 60 + iy) * 60 + ix];
#pragma unroll
        for (int oc = 0; oc < 10; oc++)
          acc[oc] = fmaf(v, sw[((oc * 8 + c) * 3 + ky) * 3 + kx], acc[oc]);
      }
    }
  float* f = feats + (size_t)idx * 12;
#pragma unroll
  for (int oc = 0; oc < 10; oc++) f[oc] = fmaxf(acc[oc], 0.f);
  f[10] = (float)ox / 60.f;
  f[11] = (float)oy / 60.f;
}

// ====== q/k/v projections (K=12 -> D=64), f32 outputs for LN stats ======
__global__ void proj_kernel(const float* __restrict__ feats,
                            const float* __restrict__ qw, const float* __restrict__ qb,
                            const float* __restrict__ kw, const float* __restrict__ kb,
                            const float* __restrict__ vw, const float* __restrict__ vb,
                            float* __restrict__ Pq, float* __restrict__ Pk, float* __restrict__ Pv) {
  __shared__ float f[12];
  int node = blockIdx.x;                  // b*3600+n
  int d = threadIdx.x;                    // 0..63
  if (d < 12) f[d] = feats[(size_t)node * 12 + d];
  __syncthreads();
  float aq = qb[d], ak = kb[d], av = vb[d];
#pragma unroll
  for (int c = 0; c < 12; c++) {
    float fv = f[c];
    aq = fmaf(fv, qw[d * 12 + c], aq);
    ak = fmaf(fv, kw[d * 12 + c], ak);
    av = fmaf(fv, vw[d * 12 + c], av);
  }
  size_t o = (size_t)node * 64 + d;
  Pq[o] = aq; Pk[o] = ak; Pv[o] = av;
}

// ====== LN statistics over (N,D) per batch for q/k/v ======
__global__ void ln_stats_kernel(const float* __restrict__ Pq, const float* __restrict__ Pk,
                                const float* __restrict__ Pv, float* __restrict__ stats) {
  __shared__ float s1[256], s2[256];
  int t = blockIdx.x / 8, bb = blockIdx.x % 8, tid = threadIdx.x;
  const float* src = (t == 0 ? Pq : (t == 1 ? Pk : Pv)) + (size_t)bb * NNODE * DDIM;
  float s = 0.f, ss = 0.f;
  for (int i = tid; i < NNODE * DDIM; i += 256) { float v = src[i]; s += v; ss = fmaf(v, v, ss); }
  s1[tid] = s; s2[tid] = ss; __syncthreads();
  for (int o = 128; o > 0; o >>= 1) {
    if (tid < o) { s1[tid] += s1[tid + o]; s2[tid] += s2[tid + o]; }
    __syncthreads();
  }
  if (tid == 0) {
    float inv = 1.f / (float)(NNODE * DDIM);
    float m = s1[0] * inv;
    float var = s2[0] * inv - m * m;
    stats[(t * 8 + bb) * 2]     = m;
    stats[(t * 8 + bb) * 2 + 1] = rsqrtf(var + EPSV);
  }
}

// ====== normalize+affine; emit Acat=[Q|K] bf16 [B,N,128] and V^T bf16 [B,64,NPAD] ======
__global__ void apply_norm_kernel(const float* __restrict__ Pq, const float* __restrict__ Pk,
                                  const float* __restrict__ Pv, const float* __restrict__ stats,
                                  const float* __restrict__ qg, const float* __restrict__ qb,
                                  const float* __restrict__ kg, const float* __restrict__ kb,
                                  const float* __restrict__ vg, const float* __restrict__ vb,
                                  unsigned short* __restrict__ Acat, unsigned short* __restrict__ Vt) {
  int node = blockIdx.x, d = threadIdx.x;
  int bb = node / NNODE, n = node % NNODE;
  size_t o = (size_t)node * 64 + d;
  int gi = n * 64 + d;                   // NH=1 -> [0,n,d]
  float mq = stats[(0 * 8 + bb) * 2], rq = stats[(0 * 8 + bb) * 2 + 1];
  float mk = stats[(1 * 8 + bb) * 2], rk = stats[(1 * 8 + bb) * 2 + 1];
  float mv = stats[(2 * 8 + bb) * 2], rv = stats[(2 * 8 + bb) * 2 + 1];
  float q = (Pq[o] - mq) * rq * qg[gi] + qb[gi];
  float k = (Pk[o] - mk) * rk * kg[gi] + kb[gi];
  float v = (Pv[o] - mv) * rv * vg[gi] + vb[gi];
  unsigned short* ac = Acat + (size_t)node * KCAT;
  ac[d]      = f2bf(q);
  ac[64 + d] = f2bf(k);
  Vt[((size_t)bb * 64 + d) * NPAD + n] = f2bf(v);
}

__global__ void vt_pad_kernel(unsigned short* __restrict__ Vt) {
  int i = blockIdx.x * 256 + threadIdx.x;       // 8*64*16 pad cols
  if (i >= BB * 64 * 16) return;
  int bd = i / 16, n = NNODE + (i % 16);
  Vt[(size_t)bd * NPAD + n] = 0;
}

// ====== pack [q_lin_w|k_lin_w] -> bf16 [NPAD,128] + combined bias ======
__global__ void pack_wcat_kernel(const float* __restrict__ qlw, const float* __restrict__ qlb,
                                 const float* __restrict__ klw, const float* __restrict__ klb,
                                 unsigned short* __restrict__ Wcat, float* __restrict__ bcat) {
  int i = blockIdx.x * 256 + threadIdx.x;
  if (i >= NPAD * KCAT) return;
  int j = i / KCAT, c = i % KCAT;
  float v = 0.f;
  if (j < NNODE) v = (c < 64) ? qlw[j * 64 + c] : klw[j * 64 + (c - 64)];
  Wcat[i] = f2bf(v);
  if (c == 0) bcat[j] = (j < NNODE) ? qlb[j] + klb[j] : 0.f;
}

// ====== pack a_lin_w -> bf16 [NPAD,NPAD] (zero-padded) + bias ======
__global__ void pack_aw_kernel(const float* __restrict__ aw, const float* __restrict__ ab,
                               unsigned short* __restrict__ AW, float* __restrict__ abp) {
  size_t i = (size_t)blockIdx.x * 256 + threadIdx.x;
  if (i >= (size_t)NPAD * NPAD) return;
  int j = (int)(i / NPAD), k = (int)(i % NPAD);
  float v = (j < NNODE && k < NNODE) ? aw[(size_t)j * NNODE + k] : 0.f;
  AW[i] = f2bf(v);
  if (k == 0) abp[j] = (j < NNODE) ? ab[j] : 0.f;
}

// ---------- WMMA fragment loads from a row-major [16 rows][32 cols] bf16 LDS tile ----------
// A-matrix layout (ISA 7.12.2): lanes 0-15 row M=L hold K=0..7,16..23; lanes 16-31 hold K=8..15,24..31
__device__ __forceinline__ v16bf load_frag_a(const unsigned int* sh, int rowBase, int lane) {
  int r = rowBase + (lane & 15);
  int dwoff = (lane & 16) ? 4 : 0;                 // K offset 8 bf16 = 4 dwords
  const unsigned int* p = sh + r * 16 + dwoff;
  union { v16bf v; u32x4 q[2]; } u;
  u.q[0] = *(const u32x4*)(p);                      // K pairs {0,1..6,7}+koff
  u.q[1] = *(const u32x4*)(p + 8);                  // K pairs {16..23}+koff
  return u.v;
}
// B-matrix layout: lane n = L&15; lanes 0-15 hold K=0..7,16..23; lanes 16-31 hold K=8..15,24..31
__device__ __forceinline__ v16bf load_frag_b(const unsigned int* sh, int rowBase, int lane) {
  int r = rowBase + (lane & 15);
  int dwoff = (lane & 16) ? 4 : 0;
  const unsigned int* p = sh + r * 16 + dwoff;
  union { v16bf v; u32x4 q[2]; } u;
  u.q[0] = *(const u32x4*)(p);
  u.q[1] = *(const u32x4*)(p + 8);
  return u.v;
}

// ====== batched bf16 GEMM: Out[b,i,j] = sum_k A[b,i,k]*W[b,j,k] (+bias[j]) ======
// block tile 128x128, 8 waves in 4x2, each wave 32(M)x64(N) = 2x4 wmma tiles, K staged 32/iter
template <int EPI /*0 none, 1 elu*/, int OUTBF /*1 bf16, 0 f32*/>
__global__ void __launch_bounds__(256)
gemm_bf16_kernel(const unsigned short* __restrict__ A, long long strideA, int lda,
                 const unsigned short* __restrict__ W, long long strideW, int ldw,
                 const float* __restrict__ bias,
                 void* __restrict__ Out, long long strideO, int ldo,
                 int M, int J, int K) {
  __shared__ unsigned int shA[128 * 16];   // 128 rows x 32 bf16
  __shared__ unsigned int shW[128 * 16];
  int tid = threadIdx.x;
  int lane = tid & 31, wv = tid >> 5;
  int wm = wv >> 1, wn = wv & 1;
  int bz = blockIdx.z;
  int m0 = blockIdx.y * 128, n0 = blockIdx.x * 128;
  const unsigned short* Ab = A + (size_t)bz * (size_t)strideA;
  const unsigned short* Wb = W + (size_t)bz * (size_t)strideW;

  // each thread stages 2x16B of A and 2x16B of W per K-step; clamp rows to keep EXEC full
  int r0 = tid >> 2, seg = tid & 3;
  int r1 = r0 + 64;
  int arow0 = min(m0 + r0, M - 1), arow1 = min(m0 + r1, M - 1);
  int wrow0 = min(n0 + r0, J - 1), wrow1 = min(n0 + r1, J - 1);
  const unsigned short* pa0 = Ab + (size_t)arow0 * lda + seg * 8;
  const unsigned short* pa1 = Ab + (size_t)arow1 * lda + seg * 8;
  const unsigned short* pw0 = Wb + (size_t)wrow0 * ldw + seg * 8;
  const unsigned short* pw1 = Wb + (size_t)wrow1 * ldw + seg * 8;

  const v8f vzero = {0.f, 0.f, 0.f, 0.f, 0.f, 0.f, 0.f, 0.f};
  v8f acc[2][4];
#pragma unroll
  for (int i = 0; i < 2; i++)
#pragma unroll
    for (int j = 0; j < 4; j++) acc[i][j] = vzero;

  int nk = K >> 5;
  for (int kk = 0; kk < nk; ++kk) {
    int k0 = kk << 5;
    u32x4 va0 = *(const u32x4*)(pa0 + k0);
    u32x4 va1 = *(const u32x4*)(pa1 + k0);
    u32x4 vw0 = *(const u32x4*)(pw0 + k0);
    u32x4 vw1 = *(const u32x4*)(pw1 + k0);
    if (kk + 1 < nk) {                               // -> global_prefetch_b8
      __builtin_prefetch(pa0 + k0 + 32, 0, 3);
      __builtin_prefetch(pw0 + k0 + 32, 0, 3);
    }
    __syncthreads();
    *(u32x4*)(shA + r0 * 16 + seg * 4) = va0;
    *(u32x4*)(shA + r1 * 16 + seg * 4) = va1;
    *(u32x4*)(shW + r0 * 16 + seg * 4) = vw0;
    *(u32x4*)(shW + r1 * 16 + seg * 4) = vw1;
    __syncthreads();

    v16bf af[2], wf[4];
#pragma unroll
    for (int i = 0; i < 2; i++) af[i] = load_frag_a(shA, wm * 32 + i * 16, lane);
#pragma unroll
    for (int j = 0; j < 4; j++) wf[j] = load_frag_b(shW, wn * 64 + j * 16, lane);
#pragma unroll
    for (int i = 0; i < 2; i++)
#pragma unroll
      for (int j = 0; j < 4; j++)
        acc[i][j] = __builtin_amdgcn_wmma_f32_16x16x32_bf16(
            false, af[i], false, wf[j], (short)0, acc[i][j], false, false);
  }

  // C/D layout: VGPR p -> M = p (+8 for lanes 16-31), N = lane&15
  int mofs = (lane & 16) ? 8 : 0;
  int ncol = lane & 15;
#pragma unroll
  for (int i = 0; i < 2; i++) {
    int mbase = m0 + wm * 32 + i * 16 + mofs;
#pragma unroll
    for (int j = 0; j < 4; j++) {
      int nn2 = n0 + wn * 64 + j * 16 + ncol;
      if (nn2 >= J) continue;
      float bv = bias ? bias[nn2] : 0.f;
#pragma unroll
      for (int p = 0; p < 8; p++) {
        int mm = mbase + p;
        if (mm >= M) continue;
        float xv = acc[i][j][p] + bv;
        if (EPI == 1) xv = eluf(xv);
        if (OUTBF)
          ((unsigned short*)Out)[(size_t)bz * (size_t)strideO + (size_t)mm * ldo + nn2] = f2bf(xv);
        else
          ((float*)Out)[(size_t)bz * (size_t)strideO + (size_t)mm * ldo + nn2] = xv;
      }
    }
  }
}

// ====== row softmax in place over bf16 logits [rows=B*N][NPAD]; pad cols -> 0 ======
__global__ void softmax_kernel(unsigned short* __restrict__ A2) {
  __shared__ float red[256];
  int row = blockIdx.x, tid = threadIdx.x;
  unsigned short* ptr = A2 + (size_t)row * NPAD;
  float m = -3.4e38f;
  for (int j = tid; j < NNODE; j += 256) m = fmaxf(m, bf2f(ptr[j]));
  red[tid] = m; __syncthreads();
  for (int o = 128; o > 0; o >>= 1) { if (tid < o) red[tid] = fmaxf(red[tid], red[tid + o]); __syncthreads(); }
  m = red[0]; __syncthreads();
  float s = 0.f;
  for (int j = tid; j < NNODE; j += 256) s += __expf(bf2f(ptr[j]) - m);
  red[tid] = s; __syncthreads();
  for (int o = 128; o > 0; o >>= 1) { if (tid < o) red[tid] += red[tid + o]; __syncthreads(); }
  float inv = 1.f / red[0];
  for (int j = tid; j < NPAD; j += 256)
    ptr[j] = (j < NNODE) ? f2bf(__expf(bf2f(ptr[j]) - m) * inv) : (unsigned short)0;
}

// ====== lin1 + relu: [B*N,64] @ [64,64]^T ======
__global__ void lin1_kernel(const float* __restrict__ E, const float* __restrict__ w,
                            const float* __restrict__ b, float* __restrict__ out) {
  __shared__ float e[64];
  int row = blockIdx.x, d = threadIdx.x;
  e[d] = E[(size_t)row * 64 + d];
  __syncthreads();
  float acc = b[d];
#pragma unroll
  for (int k2 = 0; k2 < 64; k2++) acc = fmaf(w[d * 64 + k2], e[k2], acc);
  out[(size_t)row * 64 + d] = fmaxf(acc, 0.f);
}

// ====== per-batch LN stats over (N,D) + per-d max over N ======
__global__ void final_stats_kernel(const float* __restrict__ L1, float* __restrict__ dmax,
                                   float* __restrict__ bstats) {
  __shared__ float smx[256], s1[256], s2[256];
  int bb = blockIdx.x, tid = threadIdx.x;
  int d = tid & 63, g = tid >> 6;          // 4 threads per d
  float mx = -3.4e38f, s = 0.f, ss = 0.f;
  const float* base = L1 + (size_t)bb * NNODE * 64;
  for (int n = g; n < NNODE; n += 4) {
    float v = base[(size_t)n * 64 + d];
    mx = fmaxf(mx, v); s += v; ss = fmaf(v, v, ss);
  }
  smx[tid] = mx; s1[tid] = s; s2[tid] = ss; __syncthreads();
  for (int o = 128; o > 0; o >>= 1) {
    if (tid < o) { s1[tid] += s1[tid + o]; s2[tid] += s2[tid + o]; }
    __syncthreads();
  }
  if (tid < 64)
    dmax[bb * 64 + tid] = fmaxf(fmaxf(smx[tid], smx[tid + 64]),
                                fmaxf(smx[tid + 128], smx[tid + 192]));
  if (tid == 0) {
    float inv = 1.f / (float)(NNODE * 64);
    float mean = s1[0] * inv;
    float var = s2[0] * inv - mean * mean;
    bstats[bb * 2] = mean;
    bstats[bb * 2 + 1] = rsqrtf(var + EPSV);
  }
}

// ====== head: normalize maxed features (LN is monotone per batch), lin2 + elu ======
__global__ void head_kernel(const float* __restrict__ dmax, const float* __restrict__ bstats,
                            const float* __restrict__ w2, const float* __restrict__ b2,
                            float* __restrict__ out) {
  int t = threadIdx.x;
  if (t >= 32) return;
  int bb = t >> 2, o = t & 3;
  float m = bstats[bb * 2], rs = bstats[bb * 2 + 1];
  float acc = b2[o];
#pragma unroll
  for (int d2 = 0; d2 < 64; d2++)
    acc = fmaf(w2[o * 64 + d2], (dmax[bb * 64 + d2] - m) * rs, acc);
  out[bb * 4 + o] = eluf(acc);
}

extern "C" void kernel_launch(void* const* d_in, const int* in_sizes, int n_in,
                              void* d_out, int out_size, void* d_ws, size_t ws_size,
                              hipStream_t stream) {
  const float* x   = (const float*)d_in[0];
  const float* c1w = (const float*)d_in[1];
  const float* c1b = (const float*)d_in[2];
  const float* c2w = (const float*)d_in[3];
  const float* c2b = (const float*)d_in[4];
  const float* kpw = (const float*)d_in[5];
  const float* kpb = (const float*)d_in[6];
  const float* qpw = (const float*)d_in[7];
  const float* qpb = (const float*)d_in[8];
  const float* vpw = (const float*)d_in[9];
  const float* vpb = (const float*)d_in[10];
  const float* kng = (const float*)d_in[11];
  const float* knb = (const float*)d_in[12];
  const float* qng = (const float*)d_in[13];
  const float* qnb = (const float*)d_in[14];
  const float* vng = (const float*)d_in[15];
  const float* vnb = (const float*)d_in[16];
  const float* klw = (const float*)d_in[17];
  const float* klb = (const float*)d_in[18];
  const float* qlw = (const float*)d_in[19];
  const float* qlb = (const float*)d_in[20];
  const float* alw = (const float*)d_in[21];
  const float* alb = (const float*)d_in[22];
  const float* l1w = (const float*)d_in[23];
  const float* l1b = (const float*)d_in[24];
  const float* l2w = (const float*)d_in[25];
  const float* l2b = (const float*)d_in[26];
  float* out = (float*)d_out;

  char* wsp = (char*)d_ws;
  size_t off = 0;
  auto alloc = [&](size_t bytes) -> char* {
    char* p = wsp + off;
    off = (off + bytes + 255) & ~(size_t)255;
    return p;
  };
  float* conv1out       = (float*)alloc((size_t)BB * 8 * 60 * 60 * 4);
  float* feats          = (float*)alloc((size_t)BB * NNODE * 12 * 4);
  float* Pq             = (float*)alloc((size_t)BB * NNODE * 64 * 4);
  float* Pk             = (float*)alloc((size_t)BB * NNODE * 64 * 4);
  float* Pv             = (float*)alloc((size_t)BB * NNODE * 64 * 4);
  float* stats          = (float*)alloc(24 * 2 * 4);
  unsigned short* Acat  = (unsigned short*)alloc((size_t)BB * NNODE * KCAT * 2);
  unsigned short* Vt    = (unsigned short*)alloc((size_t)BB * 64 * NPAD * 2);
  unsigned short* Wcat  = (unsigned short*)alloc((size_t)NPAD * KCAT * 2);
  float* bcat           = (float*)alloc((size_t)NPAD * 4);
  unsigned short* AW    = (unsigned short*)alloc((size_t)NPAD * NPAD * 2);
  float* abp            = (float*)alloc((size_t)NPAD * 4);
  unsigned short* A0    = (unsigned short*)alloc((size_t)BB * NNODE * NPAD * 2);
  unsigned short* A2    = (unsigned short*)alloc((size_t)BB * NNODE * NPAD * 2);
  float* Eout           = (float*)alloc((size_t)BB * NNODE * 64 * 4);
  float* L1out          = (float*)alloc((size_t)BB * NNODE * 64 * 4);
  float* dmax           = (float*)alloc(BB * 64 * 4);
  float* bstats         = (float*)alloc(BB * 2 * 4);

  conv1_kernel<<<(BB * 3600 + 255) / 256, 256, 0, stream>>>(x, c1w, c1b, conv1out);
  conv2_feats_kernel<<<(BB * 3600 + 255) / 256, 256, 0, stream>>>(conv1out, c2w, c2b, feats);
  proj_kernel<<<BB * NNODE, 64, 0, stream>>>(feats, qpw, qpb, kpw, kpb, vpw, vpb, Pq, Pk, Pv);
  ln_stats_kernel<<<24, 256, 0, stream>>>(Pq, Pk, Pv, stats);
  apply_norm_kernel<<<BB * NNODE, 64, 0, stream>>>(Pq, Pk, Pv, stats, qng, qnb, kng, knb, vng, vnb, Acat, Vt);
  vt_pad_kernel<<<(BB * 64 * 16 + 255) / 256, 256, 0, stream>>>(Vt);
  pack_wcat_kernel<<<(NPAD * KCAT + 255) / 256, 256, 0, stream>>>(qlw, qlb, klw, klb, Wcat, bcat);
  {
    size_t tot = (size_t)NPAD * NPAD;
    pack_aw_kernel<<<(unsigned)((tot + 255) / 256), 256, 0, stream>>>(alw, alb, AW, abp);
  }

  dim3 gBig((NPAD + 127) / 128, (NNODE + 127) / 128, BB);
  // GEMM1: elu(Q@qlw^T + K@klw^T + qb + kb) -> A0 bf16 (fused via K=128 concat)
  gemm_bf16_kernel<1, 1><<<gBig, 256, 0, stream>>>(
      Acat, (long long)NNODE * KCAT, KCAT, Wcat, 0, KCAT, bcat,
      A0, (long long)NNODE * NPAD, NPAD, NNODE, NPAD, KCAT);
  // GEMM2: A0 @ a_lin_w^T + a_lin_b -> A2 bf16 logits   (dominant: 7.5e11 FLOPs)
  gemm_bf16_kernel<0, 1><<<gBig, 256, 0, stream>>>(
      A0, (long long)NNODE * NPAD, NPAD, AW, 0, NPAD, abp,
      A2, (long long)NNODE * NPAD, NPAD, NNODE, NPAD, NPAD);
  softmax_kernel<<<BB * NNODE, 256, 0, stream>>>(A2);
  // GEMM3: softmax(A) @ V -> E f32
  dim3 gAV(1, (NNODE + 127) / 128, BB);
  gemm_bf16_kernel<0, 0><<<gAV, 256, 0, stream>>>(
      A2, (long long)NNODE * NPAD, NPAD, Vt, (long long)64 * NPAD, NPAD, nullptr,
      Eout, (long long)NNODE * 64, 64, NNODE, 64, NPAD);
  lin1_kernel<<<BB * NNODE, 64, 0, stream>>>(Eout, l1w, l1b, L1out);
  final_stats_kernel<<<BB, 256, 0, stream>>>(L1out, dmax, bstats);
  head_kernel<<<1, 32, 0, stream>>>(dmax, bstats, l2w, l2b, out);

  (void)in_sizes; (void)n_in; (void)out_size; (void)ws_size;
}
